// GRU_BN_single_343597384300
// MI455X (gfx1250) — compile-verified, matrix operand order checked by reading, not run on hardware
//
#include <hip/hip_runtime.h>
#include <hip/hip_bf16.h>

typedef __attribute__((ext_vector_type(16))) __bf16 v16bf;
typedef __attribute__((ext_vector_type(8)))  __bf16 v8bf;
typedef __attribute__((ext_vector_type(8)))  float  v8f;

#define HID   1024
#define BATCH 512
#define SEQ   64
#define FEAT  64

// LDS tile: 64 A-rows + 192 B-rows, 32 K-elements each, rows padded to 40
// bf16 (80 B) so fragment reads stride 20 banks (conflict-friendly).
#define ROWPAD 40
#define SBUF   (256 * ROWPAD)   // elements per buffer (A rows 0..63, B rows 64..255)

// ---- CDNA5 async global->LDS path (guarded; sync fallback) ----------------
#if defined(__has_builtin)
# if __has_builtin(__builtin_amdgcn_global_load_async_to_lds_b128)
#  define USE_ASYNC_LDS 1
# endif
#endif
#ifndef USE_ASYNC_LDS
# define USE_ASYNC_LDS 0
#endif

#if USE_ASYNC_LDS
// Builtin signature (from hipcc diagnostic): param0 = int4* in AS1 (global),
// param1 = int4* in AS3 (LDS), then two constant ints (offset, cpol).
typedef int v4i __attribute__((vector_size(16)));
typedef __attribute__((address_space(1))) v4i as1_v4i;
typedef __attribute__((address_space(3))) v4i as3_v4i;
# define ASYNC_CP16(gsrc, ldst) \
    __builtin_amdgcn_global_load_async_to_lds_b128((as1_v4i*)(gsrc), (as3_v4i*)(ldst), 0, 0)
# if __has_builtin(__builtin_amdgcn_s_wait_asynccnt)
#  define WAIT_ASYNC(n) __builtin_amdgcn_s_wait_asynccnt(n)
# else
#  define WAIT_ASYNC(n) asm volatile("s_wait_asynccnt " #n ::: "memory")
# endif
#endif

static __device__ __forceinline__ __bf16 f2bf(float f) {
    unsigned u = __float_as_uint(f);
    unsigned r = u + 0x7FFFu + ((u >> 16) & 1u);       // round-to-nearest-even
    unsigned short h = (unsigned short)(r >> 16);
    __bf16 out;
    __builtin_memcpy(&out, &h, 2);
    return out;
}

// WMMA 16x32 bf16 fragment from an LDS tile row set (ISA 7.12.2 layout:
// lane L -> row L&15, K chunks {8*sel..}, {16+8*sel..}).
static __device__ __forceinline__ v16bf lds_frag(const __bf16* __restrict__ buf,
                                                 int row_base, int lane) {
    const int row = row_base + (lane & 15);
    const int sel = lane >> 4;
    const __bf16* p = buf + row * ROWPAD + sel * 8;
    v8bf lo = *(const v8bf*)(p);
    v8bf hi = *(const v8bf*)(p + 16);
    return __builtin_shufflevector(lo, hi, 0,1,2,3,4,5,6,7,8,9,10,11,12,13,14,15);
}

static __device__ __forceinline__ v8f wmma_bf16(v16bf a, v16bf b, v8f c) {
    return __builtin_amdgcn_wmma_f32_16x16x32_bf16(false, a, false, b, (short)0, c, false, false);
}

// Stage one k-slice (A: 64x32 from Asrc, B: 192x32 gate rows from Wsrc) into
// an LDS buffer. 256 threads x 4 16-byte chunks. q==0 -> A, q>0 -> B (uniform).
static __device__ __forceinline__ void stage_tile(
    int kk, int Kx,
    const __bf16* __restrict__ X, int ldx, const __bf16* __restrict__ Wih,
    const __bf16* __restrict__ Hprev, const __bf16* __restrict__ Whh,
    int m_blk, int n_blk, __bf16* __restrict__ buf, int tid)
{
    const int  kg   = kk * 32;
    const bool xin  = (kg < Kx);
    const __bf16* Asrc = xin ? X : Hprev;
    const int     ldA  = xin ? ldx : HID;
    const __bf16* Wsrc = xin ? Wih : Whh;
    const int     ldW  = xin ? Kx : HID;
    const int     kloc = xin ? kg : (kg - Kx);
#pragma unroll
    for (int q = 0; q < 4; ++q) {
        const int c = tid + q * 256;
        const __bf16* g;
        __bf16* l;
        if (q == 0) {                       // A chunk
            const int ar = c >> 2, ch = c & 3;
            g = Asrc + (size_t)(m_blk + ar) * (size_t)ldA + (kloc + ch * 8);
            l = buf + (ar * ROWPAD + ch * 8);
        } else {                            // B chunk
            const int b = c - 256, cr = b >> 2, ch = b & 3;
            g = Wsrc + (size_t)(((cr >> 6) << 10) + n_blk + (cr & 63)) * (size_t)ldW
                     + (kloc + ch * 8);
            l = buf + ((64 + cr) * ROWPAD + ch * 8);
        }
#if USE_ASYNC_LDS
        ASYNC_CP16(g, l);
#else
        *(v8bf*)l = *(const v8bf*)g;
#endif
    }
}

// Device-wide sense-reversing barrier (all 128 blocks co-resident).
static __device__ __forceinline__ void grid_sync(unsigned* cnt, unsigned* gen, unsigned nblk) {
    __threadfence();
    __syncthreads();
    if (threadIdx.x == 0) {
        unsigned g = __hip_atomic_load(gen, __ATOMIC_ACQUIRE, __HIP_MEMORY_SCOPE_AGENT);
        unsigned p = __hip_atomic_fetch_add(cnt, 1u, __ATOMIC_ACQ_REL, __HIP_MEMORY_SCOPE_AGENT);
        if (p == nblk - 1) {
            __hip_atomic_store(cnt, 0u, __ATOMIC_RELAXED, __HIP_MEMORY_SCOPE_AGENT);
            __hip_atomic_fetch_add(gen, 1u, __ATOMIC_ACQ_REL, __HIP_MEMORY_SCOPE_AGENT);
        } else {
            while (__hip_atomic_load(gen, __ATOMIC_ACQUIRE, __HIP_MEMORY_SCOPE_AGENT) == g)
                __builtin_amdgcn_s_sleep(2);
        }
    }
    __syncthreads();
}

// One GRU step for one layer: this block's (64 x 64) tile of h_new.
static __device__ __forceinline__ void gru_phase(
    const __bf16* __restrict__ X, int ldx, int Kx,
    const __bf16* __restrict__ Wih, const __bf16* __restrict__ Whh,
    const float* __restrict__ bih, const float* __restrict__ bhh,
    const __bf16* __restrict__ Hprev, float* __restrict__ Hf32,
    __bf16* __restrict__ Hcur, __bf16* __restrict__ smem)
{
    const int tid  = threadIdx.x;
    const int lane = tid & 31;
    const int wid  = tid >> 5;
    const int wm   = wid & 3;
    const int wn   = wid >> 2;
    const int wn32 = wn * 32;
    const int m_blk = blockIdx.x * 64;
    const int n_blk = blockIdx.y * 64;
    const int NK = (Kx + HID) / 32;

    const v8f vz = {0.f,0.f,0.f,0.f,0.f,0.f,0.f,0.f};
    v8f accR[2]  = {vz, vz};
    v8f accZ[2]  = {vz, vz};
    v8f accNx[2] = {vz, vz};   // x-side of n gate
    v8f accNh[2] = {vz, vz};   // h-side of n gate (scaled by r)

    stage_tile(0, Kx, X, ldx, Wih, Hprev, Whh, m_blk, n_blk, smem, tid);
    for (int kk = 0; kk < NK; ++kk) {
        __bf16* cur = smem + (kk & 1) * SBUF;
        __bf16* nxt = smem + ((kk + 1) & 1) * SBUF;
        if (kk + 1 < NK)
            stage_tile(kk + 1, Kx, X, ldx, Wih, Hprev, Whh, m_blk, n_blk, nxt, tid);
#if USE_ASYNC_LDS
        if (kk + 1 < NK) { WAIT_ASYNC(4); } else { WAIT_ASYNC(0); }
#endif
        __syncthreads();

        const bool xin = (kk * 32) < Kx;
        v16bf a  = lds_frag(cur, wm * 16, lane);
        v16bf b0 = lds_frag(cur, 64 + 0   + wn32, lane);
        v16bf b1 = lds_frag(cur, 64 + 16  + wn32, lane);
        v16bf b2 = lds_frag(cur, 64 + 64  + wn32, lane);
        v16bf b3 = lds_frag(cur, 64 + 80  + wn32, lane);
        v16bf b4 = lds_frag(cur, 64 + 128 + wn32, lane);
        v16bf b5 = lds_frag(cur, 64 + 144 + wn32, lane);
        accR[0] = wmma_bf16(a, b0, accR[0]);
        accR[1] = wmma_bf16(a, b1, accR[1]);
        accZ[0] = wmma_bf16(a, b2, accZ[0]);
        accZ[1] = wmma_bf16(a, b3, accZ[1]);
        if (xin) {
            accNx[0] = wmma_bf16(a, b4, accNx[0]);
            accNx[1] = wmma_bf16(a, b5, accNx[1]);
        } else {
            accNh[0] = wmma_bf16(a, b4, accNh[0]);
            accNh[1] = wmma_bf16(a, b5, accNh[1]);
        }
        __syncthreads();   // protect cur from being restaged next-next iter
    }

    // ---- gate epilogue ----
    const int col = lane & 15;
    const int sel = lane >> 4;
#pragma unroll
    for (int ns = 0; ns < 2; ++ns) {
#pragma unroll
        for (int e = 0; e < 8; ++e) {
            const int m = m_blk + wm * 16 + e + 8 * sel;
            const int n = n_blk + wn32 + ns * 16 + col;
            float r  = accR[ns][e]  + bih[n]        + bhh[n];
            float z  = accZ[ns][e]  + bih[n + 1024] + bhh[n + 1024];
            float xn = accNx[ns][e] + bih[n + 2048];
            float hn = accNh[ns][e] + bhh[n + 2048];
            r = 1.f / (1.f + __expf(-r));
            z = 1.f / (1.f + __expf(-z));
            float nn = tanhf(xn + r * hn);
            const size_t idx = (size_t)m * HID + n;
            float hp = Hf32[idx];
            float hv = (1.f - z) * nn + z * hp;
            Hf32[idx] = hv;
            Hcur[idx] = f2bf(hv);
        }
    }
}

struct GruP {
    const __bf16* xbf;
    const __bf16* wih[3];
    const __bf16* whh[3];
    const float*  bih[3];
    const float*  bhh[3];
    float*        hf32[3];
    __bf16*       hbf[3][2];
    unsigned*     cnt;
    unsigned*     gen;
};

// Persistent kernel: all 64 time steps x 3 layers, grid barrier between phases.
__global__ __launch_bounds__(256)
void gru_persist_k(GruP P) {
    __shared__ __bf16 smem[2 * SBUF];   // 40 KB, double-buffered tiles
    const unsigned nblk = gridDim.x * gridDim.y;
    for (int t = 0; t < SEQ; ++t) {
        const int pb = t & 1;       // prev-h buffer (zeroed before t=0)
        const int cb = pb ^ 1;      // new-h buffer
        gru_phase(P.xbf + (size_t)t * FEAT, SEQ * FEAT, FEAT,
                  P.wih[0], P.whh[0], P.bih[0], P.bhh[0],
                  P.hbf[0][pb], P.hf32[0], P.hbf[0][cb], smem);
        grid_sync(P.cnt, P.gen, nblk);
        gru_phase(P.hbf[0][cb], HID, HID,
                  P.wih[1], P.whh[1], P.bih[1], P.bhh[1],
                  P.hbf[1][pb], P.hf32[1], P.hbf[1][cb], smem);
        grid_sync(P.cnt, P.gen, nblk);
        gru_phase(P.hbf[1][cb], HID, HID,
                  P.wih[2], P.whh[2], P.bih[2], P.bhh[2],
                  P.hbf[2][pb], P.hf32[2], P.hbf[2][cb], smem);
        grid_sync(P.cnt, P.gen, nblk);
    }
}

// ---------------------------------------------------------------------------
__global__ void f32_to_bf16_k(const float* __restrict__ in, __bf16* __restrict__ out, int n) {
    int i = blockIdx.x * blockDim.x + threadIdx.x;
    for (; i < n; i += gridDim.x * blockDim.x) out[i] = f2bf(in[i]);
}

__global__ void zero_u32_k(unsigned* __restrict__ p, int n) {
    int i = blockIdx.x * blockDim.x + threadIdx.x;
    for (; i < n; i += gridDim.x * blockDim.x) p[i] = 0u;
}

__global__ __launch_bounds__(256)
void bn_stats_k(const float* __restrict__ h, float* __restrict__ mu, float* __restrict__ rs) {
    __shared__ float s1[256], s2[256];
    const int n = blockIdx.x, tid = threadIdx.x;
    float a = 0.f, b = 0.f;
    for (int m = tid; m < BATCH; m += 256) {
        float v = h[(size_t)m * HID + n];
        a += v; b += v * v;
    }
    s1[tid] = a; s2[tid] = b; __syncthreads();
    for (int s = 128; s > 0; s >>= 1) {
        if (tid < s) { s1[tid] += s1[tid + s]; s2[tid] += s2[tid + s]; }
        __syncthreads();
    }
    if (tid == 0) {
        float m_ = s1[0] / (float)BATCH;
        float v_ = s2[0] / (float)BATCH - m_ * m_;
        mu[n] = m_;
        rs[n] = rsqrtf(v_ + 1e-5f);
    }
}

__global__ __launch_bounds__(256)
void fc_k(const float* __restrict__ h, const float* __restrict__ mu, const float* __restrict__ rs,
          const float* __restrict__ gamma, const float* __restrict__ beta,
          const float* __restrict__ fcW, const float* __restrict__ fcb,
          float* __restrict__ out)
{
    __shared__ float s[256];
    const int b = blockIdx.x, tid = threadIdx.x;
    float acc = 0.f;
    for (int n = tid; n < HID; n += 256) {
        float v = (h[(size_t)b * HID + n] - mu[n]) * rs[n] * gamma[n] + beta[n];
        acc += v * fcW[n];
    }
    s[tid] = acc; __syncthreads();
    for (int st = 128; st > 0; st >>= 1) {
        if (tid < st) s[tid] += s[tid + st];
        __syncthreads();
    }
    if (tid == 0) out[b] = 1.f / (1.f + __expf(-(s[0] + fcb[0])));
}

// ---------------------------------------------------------------------------
extern "C" void kernel_launch(void* const* d_in, const int* in_sizes, int n_in,
                              void* d_out, int out_size, void* d_ws, size_t ws_size,
                              hipStream_t stream) {
    (void)in_sizes; (void)n_in; (void)out_size; (void)ws_size;

    const float* x      = (const float*)d_in[0];
    const float* Wih[3] = { (const float*)d_in[1], (const float*)d_in[5], (const float*)d_in[9]  };
    const float* Whh[3] = { (const float*)d_in[2], (const float*)d_in[6], (const float*)d_in[10] };
    const float* bih[3] = { (const float*)d_in[3], (const float*)d_in[7], (const float*)d_in[11] };
    const float* bhh[3] = { (const float*)d_in[4], (const float*)d_in[8], (const float*)d_in[12] };
    const float* gamma = (const float*)d_in[13];
    const float* beta  = (const float*)d_in[14];
    const float* fcW   = (const float*)d_in[15];
    const float* fcb   = (const float*)d_in[16];
    float* out = (float*)d_out;

    // ---- workspace carve-up (bump allocator, 256B aligned) ----
    char* ws = (char*)d_ws;
    size_t off = 0;
    auto alloc = [&](size_t bytes) -> void* {
        void* p = ws + off;
        off = (off + bytes + 255) & ~(size_t)255;
        return p;
    };
    GruP P;
    for (int l = 0; l < 3; ++l) P.hf32[l] = (float*)alloc((size_t)BATCH * HID * 4);
    for (int l = 0; l < 3; ++l)
        for (int p = 0; p < 2; ++p) P.hbf[l][p] = (__bf16*)alloc((size_t)BATCH * HID * 2);
    unsigned* bar = (unsigned*)alloc(2 * sizeof(unsigned));
    P.cnt = bar; P.gen = bar + 1;
    const size_t zero_words = off / 4;       // state + barrier vars

    __bf16* xbf = (__bf16*)alloc((size_t)BATCH * SEQ * FEAT * 2);
    P.xbf = xbf;
    const int KxL[3] = { FEAT, HID, HID };
    __bf16* wihb[3], *whhb[3];
    for (int l = 0; l < 3; ++l) wihb[l] = (__bf16*)alloc((size_t)3 * HID * KxL[l] * 2);
    for (int l = 0; l < 3; ++l) whhb[l] = (__bf16*)alloc((size_t)3 * HID * HID * 2);
    float* mu = (float*)alloc(HID * 4);
    float* rs = (float*)alloc(HID * 4);
    for (int l = 0; l < 3; ++l) {
        P.wih[l] = wihb[l]; P.whh[l] = whhb[l];
        P.bih[l] = bih[l];  P.bhh[l] = bhh[l];
    }

    // ---- zero the recurrent state + barrier vars (every call: determinism) ----
    zero_u32_k<<<2048, 256, 0, stream>>>((unsigned*)ws, (int)zero_words);

    // ---- f32 -> bf16 conversions ----
    f32_to_bf16_k<<<2048, 256, 0, stream>>>(x, xbf, BATCH * SEQ * FEAT);
    for (int l = 0; l < 3; ++l) {
        f32_to_bf16_k<<<2048, 256, 0, stream>>>(Wih[l], wihb[l], 3 * HID * KxL[l]);
        f32_to_bf16_k<<<2048, 256, 0, stream>>>(Whh[l], whhb[l], 3 * HID * HID);
    }

    // ---- the whole recurrence in one persistent kernel ----
    dim3 sgrid(BATCH / 64, HID / 64);   // (8, 16) = 128 blocks, all co-resident
    gru_persist_k<<<sgrid, 256, 0, stream>>>(P);

    // ---- batch-norm over batch + FC + sigmoid ----
    bn_stats_k<<<HID, 256, 0, stream>>>(P.hf32[2], mu, rs);
    fc_k<<<BATCH, 256, 0, stream>>>(P.hf32[2], mu, rs, gamma, beta, fcW, fcb, out);
}
